// GATModel_82351702934000
// MI455X (gfx1250) — compile-verified, matrix operand order checked by reading, not run on hardware
//
#include <hip/hip_runtime.h>

// ---------------------------------------------------------------------------
// GATv2 x5 + BN/ELU + edge dot, MI455X (gfx1250, wave32).
//  - Node transforms: bf16 WMMA (v_wmma_f32_16x16x32_bf16), one wave computes
//    a 16x128 strip for BOTH Wl and Wr from a single A fragment (16 WMMAs per
//    A load). Weights staged in LDS via the Tensor Data Mover
//    (tensor_load_to_lds, TENSORcnt) -- CDNA5 async-tensor path.
//  - Edge phase: L2-resident gather/softmax/scatter with f32 atomics
//    (all tensors < 192MB L2; HBM traffic ~0.3GB total => ~13us at 23.3TB/s).
// ---------------------------------------------------------------------------

#define NODES 50000
#define EDGES 800000
#define INC   64
#define HIDF  128     // H*HID = H*OUT = feature width of xl/xr/acc
#define HEADS 4
#define CH    32      // channels per head
#define EDIM  16
#define BN_EPS 1e-5f
#define SLOPE  0.2f

typedef __bf16 bf16;
typedef __attribute__((ext_vector_type(16))) __bf16 v16bf;
typedef __attribute__((ext_vector_type(8)))  float  v8f;
typedef __attribute__((ext_vector_type(4)))  unsigned tdm_g0_t;
typedef __attribute__((ext_vector_type(8)))  unsigned tdm_g1_t;

// ---------------- TDM: 1-D tensor load Global -> LDS ----------------
// D# per CDNA5 ISA. data_size=2B; tensor = one row of
// nelem elements; tile = same row. Tracked by TENSORcnt.
__device__ __forceinline__ void tdm_load_1d(const void* gptr, unsigned lds_off,
                                            unsigned nelem /* 2-byte units */) {
  unsigned long long ga = (unsigned long long)gptr;
  tdm_g0_t g0;
  g0[0] = 1u;                                   // count=1, user mode
  g0[1] = lds_off;                              // lds_addr (bytes)
  g0[2] = (unsigned)ga;                         // global_addr[31:0]
  g0[3] = (unsigned)(ga >> 32) | 0x80000000u;   // global_addr[56:32] | type=2
  tdm_g1_t g1;
  g1[0] = 0x00010000u;                          // data_size=1 (2 bytes)
  g1[1] = (nelem & 0xFFFFu) << 16;              // tensor_dim0[15:0]
  g1[2] = (nelem >> 16) | (1u << 16);           // tensor_dim0[31:16] | tensor_dim1=1
  g1[3] = (nelem & 0xFFFFu) << 16;              // tile_dim0
  g1[4] = 0u;                                   // tile_dim1=0, tile_dim2=0
  g1[5] = nelem;                                // tensor_dim0_stride[31:0]
  g1[6] = 0u;
  g1[7] = 0u;
  asm volatile("tensor_load_to_lds %0, %1" : : "s"(g0), "s"(g1) : "memory");
}

// ---------------- utility kernels ----------------

__global__ void k_fill_zero(float* __restrict__ p, long n) {
  long i = (long)blockIdx.x * blockDim.x + threadIdx.x;
  if (i < n) p[i] = 0.f;
}

__global__ void k_f32_to_bf16(const float* __restrict__ in, bf16* __restrict__ out, long n) {
  long i = (long)blockIdx.x * blockDim.x + threadIdx.x;
  if (i < n) out[i] = (bf16)in[i];
}

// W [K x 128] f32 -> Wt [128 x K] bf16 (transposed: WMMA B-fragment loads
// become contiguous per lane)
__global__ void k_wt(const float* __restrict__ W, bf16* __restrict__ Wt, int K) {
  int t = blockIdx.x * blockDim.x + threadIdx.x;
  if (t >= K * HIDF) return;
  int k = t / HIDF, n = t % HIDF;
  Wt[n * K + k] = (bf16)W[k * HIDF + n];
}

// ---------------- dual WMMA GEMM ----------------
// Cl = A*Wl, Cr = A*Wr. A [M x K] bf16 row-major; Wlt/Wrt = W^T [128 x K] bf16.
// Block = 8 waves; wave w owns M-tile (blockIdx.x*8+w); both weight matrices
// staged in LDS by the TDM; each wave computes 16 accumulator tiles (8 N-tiles
// x {l,r}) reusing one A fragment per k-step.
// A-fragment (ISA 7.12.2, 16-bit A 16x32): lanes 0-15 carry rows M=lane with
// K={k0..k0+7, k0+16..k0+23}; lanes 16-31 carry K={k0+8..k0+15, k0+24..k0+31}.
// B fragment assumed symmetric on B^T.
__global__ void __launch_bounds__(256)
k_gemm_dual(const bf16* __restrict__ A, const bf16* __restrict__ Wlt,
            const bf16* __restrict__ Wrt, float* __restrict__ Cl,
            float* __restrict__ Cr, int M, int K) {
  extern __shared__ bf16 lds[];                 // [2][128][K]
  bf16* lwl = lds;
  bf16* lwr = lds + (size_t)HIDF * K;

  if (threadIdx.x == 0) {
    unsigned base = (unsigned)(unsigned long long)(uintptr_t)lds;
    unsigned nelem = (unsigned)(HIDF * K);
    tdm_load_1d(Wlt, base, nelem);
    tdm_load_1d(Wrt, base + nelem * 2u, nelem);
    __builtin_amdgcn_s_wait_tensorcnt(0);
  }
  __syncthreads();

  const int lane  = threadIdx.x & 31;
  const int wave  = threadIdx.x >> 5;           // 0..7
  const int tileM = blockIdx.x * 8 + wave;
  if (tileM * 16 >= M) return;                  // wave-uniform exit
  const int half = lane >> 4;
  const int l16  = lane & 15;
  const long row = (long)tileM * 16 + l16;

  v8f accl[8], accr[8];
#pragma unroll
  for (int i = 0; i < 8; ++i) { accl[i] = (v8f){}; accr[i] = (v8f){}; }

  for (int k0 = 0; k0 < K; k0 += 32) {
    v16bf a;
    const bf16* ap = A + row * K + k0 + half * 8;
#pragma unroll
    for (int j = 0; j < 8; ++j) { a[j] = ap[j]; a[j + 8] = ap[16 + j]; }
#pragma unroll
    for (int nt = 0; nt < 8; ++nt) {
      const bf16* blp = lwl + (long)(nt * 16 + l16) * K + k0 + half * 8;
      const bf16* brp = lwr + (long)(nt * 16 + l16) * K + k0 + half * 8;
      v16bf bl, br;
#pragma unroll
      for (int j = 0; j < 8; ++j) {
        bl[j] = blp[j]; bl[j + 8] = blp[16 + j];
        br[j] = brp[j]; br[j + 8] = brp[16 + j];
      }
      accl[nt] = __builtin_amdgcn_wmma_f32_16x16x32_bf16(
          false, a, false, bl, (short)0, accl[nt], false, false);
      accr[nt] = __builtin_amdgcn_wmma_f32_16x16x32_bf16(
          false, a, false, br, (short)0, accr[nt], false, false);
    }
  }
  // C layout: VGPR j -> M = j (lanes 0-15) / M = 8+j (lanes 16-31), N = l16
#pragma unroll
  for (int nt = 0; nt < 8; ++nt) {
#pragma unroll
    for (int j = 0; j < 8; ++j) {
      long r = (long)tileM * 16 + j + half * 8;
      Cl[r * HIDF + nt * 16 + l16] = accl[nt][j];
      Cr[r * HIDF + nt * 16 + l16] = accr[nt][j];
    }
  }
}

// ---------------- ordered-uint float max encoding ----------------

__device__ __forceinline__ unsigned fenc(float x) {
  unsigned u = __float_as_uint(x);
  return (u & 0x80000000u) ? ~u : (u | 0x80000000u);
}
__device__ __forceinline__ float fdec(unsigned k) {
  return (k & 0x80000000u) ? __uint_as_float(k & 0x7fffffffu)
                           : __uint_as_float(~k);
}

// ---------------- edge pass A: logits + segment max ----------------
__global__ void k_edge_logits(const float* __restrict__ xl, const float* __restrict__ xr,
                              const float* __restrict__ ea, const float* __restrict__ We,
                              const float* __restrict__ att,
                              const int* __restrict__ src, const int* __restrict__ dst,
                              float* __restrict__ logits, unsigned* __restrict__ mx) {
  int t = blockIdx.x * blockDim.x + threadIdx.x;
  if (t >= EDGES * HEADS) return;
  int e = t >> 2, h = t & 3;
  int s = src[e], d = dst[e];
  const float* xls = xl + (long)s * HIDF + h * CH;
  const float* xrd = xr + (long)d * HIDF + h * CH;
  const float* eap = ea + (long)e * EDIM;
  float ef[EDIM];
#pragma unroll
  for (int k = 0; k < EDIM; ++k) ef[k] = eap[k];
  const float* wec = We + h * CH;
  const float* ath = att + h * CH;
  float acc = 0.f;
#pragma unroll 4
  for (int c = 0; c < CH; ++c) {
    float ev = 0.f;
#pragma unroll
    for (int k = 0; k < EDIM; ++k) ev += ef[k] * wec[k * HIDF + c];
    float m = xls[c] + xrd[c] + ev;
    m = (m > 0.f) ? m : SLOPE * m;
    acc += m * ath[c];
  }
  logits[t] = acc;
  atomicMax(mx + (long)d * HEADS + h, fenc(acc));
}

// ---------------- edge pass B: exp + segment sum ----------------
__global__ void k_edge_exp(const int* __restrict__ dst, const unsigned* __restrict__ mx,
                           float* __restrict__ logits, float* __restrict__ den) {
  int t = blockIdx.x * blockDim.x + threadIdx.x;
  if (t >= EDGES * HEADS) return;
  int e = t >> 2, h = t & 3;
  int d = dst[e];
  float ex = __expf(logits[t] - fdec(mx[(long)d * HEADS + h]));
  logits[t] = ex;
  atomicAdd(den + (long)d * HEADS + h, ex);
}

// ---------------- edge pass C: alpha + scatter ----------------
__global__ void k_edge_scatter(const float* __restrict__ xl,
                               const int* __restrict__ src, const int* __restrict__ dst,
                               const float* __restrict__ exb, const float* __restrict__ den,
                               float* __restrict__ acc, float* __restrict__ alpha_out) {
  int t = blockIdx.x * blockDim.x + threadIdx.x;
  if (t >= EDGES * HEADS) return;
  int e = t >> 2, h = t & 3;
  int s = src[e], d = dst[e];
  float alpha = exb[t] / den[(long)d * HEADS + h];
  if (alpha_out) alpha_out[t] = alpha;
  const float* xs = xl + (long)s * HIDF + h * CH;
  float* ad = acc + (long)d * HIDF + h * CH;
#pragma unroll
  for (int c = 0; c < CH; ++c) atomicAdd(ad + c, xs[c] * alpha);
}

// ---------------- BatchNorm: per-column partial sums ----------------
__global__ void k_bn_stats(const float* __restrict__ acc, const float* __restrict__ bias,
                           float* __restrict__ sum, float* __restrict__ sumsq) {
  __shared__ float s_s[256], s_q[256];
  int c    = threadIdx.x & 127;
  int half = threadIdx.x >> 7;
  int r0   = blockIdx.x * 64 + half * 32;
  float s = 0.f, q = 0.f;
  for (int i = 0; i < 32; ++i) {
    int r = r0 + i;
    if (r < NODES) {
      float v = acc[(long)r * HIDF + c] + bias[c];
      s += v; q += v * v;
    }
  }
  s_s[threadIdx.x] = s; s_q[threadIdx.x] = q;
  __syncthreads();
  if (half == 0) {
    s += s_s[threadIdx.x + 128];
    q += s_q[threadIdx.x + 128];
    atomicAdd(&sum[c], s);
    atomicAdd(&sumsq[c], q);
  }
}

__global__ void k_bn_final(const float* __restrict__ sum, const float* __restrict__ sumsq,
                           const float* __restrict__ gamma, const float* __restrict__ beta,
                           float* __restrict__ scale, float* __restrict__ shift) {
  int c = threadIdx.x;   // 128 threads
  float mu  = sum[c] * (1.f / NODES);
  float var = sumsq[c] * (1.f / NODES) - mu * mu;
  float sc  = gamma[c] * rsqrtf(var + BN_EPS);
  scale[c] = sc;
  shift[c] = beta[c] - mu * sc;
}

// BN + ELU fused with bf16 conversion feeding the next layer's GEMM.
__global__ void k_bn_apply(const float* __restrict__ acc, const float* __restrict__ bias,
                           const float* __restrict__ scale, const float* __restrict__ shift,
                           bf16* __restrict__ hb) {
  long i = (long)blockIdx.x * blockDim.x + threadIdx.x;
  if (i >= (long)NODES * HIDF) return;
  int c = (int)(i & (HIDF - 1));
  float v = acc[i] + bias[c];
  v = v * scale[c] + shift[c];
  v = (v > 0.f) ? v : (__expf(v) - 1.f);
  hb[i] = (bf16)v;
}

// ---------------- layer 5: head mean + bias ----------------
__global__ void k_head_mean(const float* __restrict__ acc, const float* __restrict__ bias,
                            float* __restrict__ h5) {
  long i = (long)blockIdx.x * blockDim.x + threadIdx.x;
  if (i >= (long)NODES * CH) return;
  long r = i / CH;
  int  c = (int)(i % CH);
  const float* a = acc + r * HIDF + c;
  h5[i] = 0.25f * (a[0] + a[32] + a[64] + a[96]) + bias[c];
}

// ---------------- final per-edge dot ----------------
__global__ void k_edge_dot(const float* __restrict__ h5, const int* __restrict__ src,
                           const int* __restrict__ dst, float* __restrict__ out) {
  int e = blockIdx.x * blockDim.x + threadIdx.x;
  if (e >= EDGES) return;
  const float* a = h5 + (long)src[e] * CH;
  const float* b = h5 + (long)dst[e] * CH;
  float s = 0.f;
#pragma unroll
  for (int c = 0; c < CH; ++c) s += a[c] * b[c];
  out[e] = s;
}

// ---------------------------------------------------------------------------

extern "C" void kernel_launch(void* const* d_in, const int* in_sizes, int n_in,
                              void* d_out, int out_size, void* d_ws, size_t ws_size,
                              hipStream_t stream) {
  (void)in_sizes; (void)n_in; (void)out_size; (void)ws_size;

  const float* x    = (const float*)d_in[0];
  const int*   eidx = (const int*)d_in[1];
  const int*   src  = eidx;
  const int*   dst  = eidx + EDGES;
  const float* ea   = (const float*)d_in[2];
  const float *Wl[5], *Wr[5], *We[5], *att[5], *bias[5];
  for (int l = 0; l < 5; ++l) {
    Wl[l]   = (const float*)d_in[3 + 5 * l + 0];
    Wr[l]   = (const float*)d_in[3 + 5 * l + 1];
    We[l]   = (const float*)d_in[3 + 5 * l + 2];
    att[l]  = (const float*)d_in[3 + 5 * l + 3];
    bias[l] = (const float*)d_in[3 + 5 * l + 4];
  }
  const float *gamma[4], *beta[4];
  for (int b = 0; b < 4; ++b) {
    gamma[b] = (const float*)d_in[28 + 2 * b + 0];
    beta[b]  = (const float*)d_in[28 + 2 * b + 1];
  }
  float* out_edge  = (float*)d_out;           // [E]
  float* out_alpha = (float*)d_out + EDGES;   // [E,H]

  // ---- carve workspace (~110 MB) ----
  char* ws = (char*)d_ws;
  size_t off = 0;
  auto carve = [&](size_t bytes) -> char* {
    char* p = ws + off;
    off += (bytes + 255) & ~(size_t)255;
    return p;
  };
  bf16*     hb      = (bf16*)carve((size_t)NODES * HIDF * sizeof(bf16));
  bf16*     wlt     = (bf16*)carve((size_t)HIDF * HIDF * sizeof(bf16));
  bf16*     wrt     = (bf16*)carve((size_t)HIDF * HIDF * sizeof(bf16));
  float*    xl      = (float*)carve((size_t)NODES * HIDF * sizeof(float));
  float*    xr      = (float*)carve((size_t)NODES * HIDF * sizeof(float));
  float*    accb    = (float*)carve((size_t)NODES * HIDF * sizeof(float));
  float*    logits  = (float*)carve((size_t)EDGES * HEADS * sizeof(float));
  unsigned* mx      = (unsigned*)carve((size_t)NODES * HEADS * sizeof(unsigned));
  float*    den     = (float*)carve((size_t)NODES * HEADS * sizeof(float));
  float*    bnsum   = (float*)carve(HIDF * sizeof(float));
  float*    bnsq    = (float*)carve(HIDF * sizeof(float));
  float*    bnscale = (float*)carve(HIDF * sizeof(float));
  float*    bnshift = (float*)carve(HIDF * sizeof(float));
  float*    h5      = (float*)carve((size_t)NODES * CH * sizeof(float));

  const dim3 b256(256);
  auto nb = [](long n) { return dim3((unsigned)((n + 255) / 256)); };
  const long EH = (long)EDGES * HEADS;

  // layer-0 activations -> bf16
  k_f32_to_bf16<<<nb((long)NODES * INC), b256, 0, stream>>>(x, hb, (long)NODES * INC);

  for (int l = 0; l < 5; ++l) {
    const int K = (l == 0) ? INC : HIDF;

    // weights -> bf16 transposed (global staging for the TDM)
    k_wt<<<nb((long)K * HIDF), b256, 0, stream>>>(Wl[l], wlt, K);
    k_wt<<<nb((long)K * HIDF), b256, 0, stream>>>(Wr[l], wrt, K);

    // node transforms via WMMA (both Wl and Wr in one pass)
    const unsigned ldsBytes = 2u * HIDF * (unsigned)K * sizeof(bf16);
    k_gemm_dual<<<dim3((NODES / 16 + 7) / 8), b256, ldsBytes, stream>>>(
        hb, wlt, wrt, xl, xr, NODES, K);

    // per-layer init: mx = 0u (== -inf key), den = 0, acc = 0
    k_fill_zero<<<nb((long)NODES * HEADS), b256, 0, stream>>>((float*)mx, (long)NODES * HEADS);
    k_fill_zero<<<nb((long)NODES * HEADS), b256, 0, stream>>>(den, (long)NODES * HEADS);
    k_fill_zero<<<nb((long)NODES * HIDF),  b256, 0, stream>>>(accb, (long)NODES * HIDF);

    // edge phase
    k_edge_logits<<<nb(EH), b256, 0, stream>>>(xl, xr, ea, We[l], att[l], src, dst, logits, mx);
    k_edge_exp<<<nb(EH), b256, 0, stream>>>(dst, mx, logits, den);
    k_edge_scatter<<<nb(EH), b256, 0, stream>>>(xl, src, dst, logits, den, accb,
                                                (l == 4) ? out_alpha : (float*)nullptr);

    if (l < 4) {
      k_fill_zero<<<1, 256, 0, stream>>>(bnsum, HIDF);
      k_fill_zero<<<1, 256, 0, stream>>>(bnsq, HIDF);
      k_bn_stats<<<dim3((NODES + 63) / 64), b256, 0, stream>>>(accb, bias[l], bnsum, bnsq);
      k_bn_final<<<1, 128, 0, stream>>>(bnsum, bnsq, gamma[l], beta[l], bnscale, bnshift);
      k_bn_apply<<<nb((long)NODES * HIDF), b256, 0, stream>>>(accb, bias[l], bnscale, bnshift, hb);
    } else {
      k_head_mean<<<nb((long)NODES * CH), b256, 0, stream>>>(accb, bias[l], h5);
      k_edge_dot<<<nb(EDGES), b256, 0, stream>>>(h5, src, dst, out_edge);
    }
  }
}
